// CouplingLayer_79164837200472
// MI455X (gfx1250) — matrix operand order repaired and checked by reading, not compiled, not run on hardware
//
#include <hip/hip_runtime.h>
#include <hip/hip_bf16.h>
#include <math.h>

typedef __attribute__((ext_vector_type(16))) _Float16 v16h;
typedef __attribute__((ext_vector_type(8)))  _Float16 v8h;
typedef __attribute__((ext_vector_type(2)))  __fp16   v2fp;   // cvt_pkrtz result type
typedef __attribute__((ext_vector_type(8)))  float    v8f;

union BOp  { v16h v; v8h h8[2]; unsigned u[8]; };
union PK2  { v2fp h; unsigned u; };

#define BIG 10000.0f
#define WAVES_PER_BLOCK 8       // 256 threads, 32 points per wave, 256/block

// ---------------------------------------------------------------------------
// Pre-pack weights into WMMA *A*-operand layout (D = A x B, A = W^T tiles).
// A 16x32 packing in this code's contiguous K order: lane l holds row M = l%16; half i -> K =
// ((l>=16)?16:0) + i.
//   Ap1: [4 hidden-tiles][32 lanes][16 halfs]; K=18 row carries b1 (bias
//        folded via a constant 1.0 feature in z), K>18 zero.
//   Ap2: [2 K-tiles][32 lanes][16 halfs]; rows M>=10 zero.
//   b2p: [16] f32, b2 padded -> direct v8f C-init load.
// ---------------------------------------------------------------------------
__global__ void pack_weights_kernel(const float* __restrict__ W1,
                                    const float* __restrict__ b1,
                                    const float* __restrict__ W2,
                                    const float* __restrict__ b2,
                                    _Float16* __restrict__ Ap1,
                                    _Float16* __restrict__ Ap2,
                                    float* __restrict__ b2p) {
  int tid = blockIdx.x * blockDim.x + threadIdx.x;
  if (tid < 4 * 32 * 16) {              // Ap1
    int t    = tid >> 9;
    int lane = (tid >> 4) & 31;
    int i    = tid & 15;
    int M    = t * 16 + (lane & 15);    // hidden unit
    int K    = ((lane >= 16) ? 16 : 0) + i;
    float v  = (K < 18) ? W1[K * 64 + M] : ((K == 18) ? b1[M] : 0.0f);
    Ap1[tid] = (_Float16)v;
  }
  if (tid < 2 * 32 * 16) {              // Ap2
    int kt   = tid >> 9;
    int lane = (tid >> 4) & 31;
    int i    = tid & 15;
    int M    = lane & 15;               // output dim (10 padded to 16)
    int K    = kt * 32 + ((lane >= 16) ? 16 : 0) + i;
    float v  = (M < 10) ? W2[K * 10 + M] : 0.0f;
    Ap2[tid] = (_Float16)v;
  }
  if (tid < 16) b2p[tid] = (tid < 10) ? b2[tid] : 0.0f;
}

__device__ __forceinline__ unsigned shx16u(unsigned v) {
  return (unsigned)__shfl_xor((int)v, 16, 32);
}
__device__ __forceinline__ float shx16f(float v) {
  return __shfl_xor(v, 16, 32);
}
__device__ __forceinline__ float fast_tanh(float v) {
  float e = __expf(-2.0f * fabsf(v));
  return copysignf(__fdividef(1.0f - e, 1.0f + e), v);
}
__device__ __forceinline__ unsigned pk16(float a, float b) {
  PK2 pk;
  pk.h = __builtin_amdgcn_cvt_pkrtz(a, b);
  return pk.u;
}

// ---------------------------------------------------------------------------
// One wave = 32 points, one point per lane. No LDS, no barriers.
// Orientation: D = W^T_tile (A) x z^T (B);  C columns = points.
// ---------------------------------------------------------------------------
__global__ __launch_bounds__(256) void coupling_wmma_kernel(
    const float* __restrict__ x,        // (N,3)
    const float* __restrict__ t_feat,   // (N,16)
    const float* __restrict__ mask,     // (3,)
    const _Float16* __restrict__ Ap1,   // packed W1^T A-operands (+bias row)
    const _Float16* __restrict__ Ap2,   // packed W2^T A-operands
    const float* __restrict__ b2p,      // padded b2
    float* __restrict__ out)            // (N,3)
{
  const int  lane = threadIdx.x & 31;
  const int  wv   = threadIdx.x >> 5;
  const int  base = (blockIdx.x * WAVES_PER_BLOCK + wv) * 32;
  const int  p    = base + lane;        // this lane's point
  const bool low  = lane < 16;

  // ---- per-lane point load -------------------------------------------------
  const float4* tf = (const float4*)(t_feat + (size_t)p * 16);
  float4 t0 = tf[0], t1 = tf[1], t2 = tf[2], t3 = tf[3];
  const float* xp = x + (size_t)p * 3;
  float x0 = xp[0], x1 = xp[1], x2 = xp[2];

  float a0 = fast_tanh(x0), a1 = fast_tanh(x1);

  // ---- pack z = [a0,a1,t0..t15,1,0...] as f16 pairs (K=0..19) -------------
  unsigned zp[9];
  zp[0] = pk16(a0, a1);
  zp[1] = pk16(t0.x, t0.y);
  zp[2] = pk16(t0.z, t0.w);
  zp[3] = pk16(t1.x, t1.y);
  zp[4] = pk16(t1.z, t1.w);
  zp[5] = pk16(t2.x, t2.y);
  zp[6] = pk16(t2.z, t2.w);
  zp[7] = pk16(t3.x, t3.y);
  zp[8] = pk16(t3.z, t3.w);             // t14,t15
  const unsigned zbias = 0x00003C00u;   // packed (f16)1.0, (f16)0.0  (K=18,19)

  // ---- B1 operands (z^T), point-tiles m=0 (points 0-15) / m=1 (16-31) -----
  // B order used here (matches A): lane = N%16 + 16*(K-half); per lane 16 consecutive K halfs.
  unsigned shz[9];
#pragma unroll
  for (int i = 0; i < 9; ++i) shz[i] = shx16u(zp[i]);
  BOp B1a, B1b;
#pragma unroll
  for (int r = 0; r < 8; ++r) {
    unsigned hiA = (r == 0) ? shz[8] : ((r == 1) ? zbias : 0u); // K16..31
    unsigned hiB = (r == 0) ? zp[8]  : ((r == 1) ? zbias : 0u);
    B1a.u[r] = low ? zp[r]  : hiA;
    B1b.u[r] = low ? shz[r] : hiB;
  }

  // ---- layer 1: h = relu(W1^T z + b1); 4 hidden-tiles x 2 point-tiles -----
  unsigned hpk[4][2][4];                // packed f16 h, [tile][m][pair]
#pragma unroll
  for (int t = 0; t < 4; ++t) {
    v16h A1 = *(const v16h*)(Ap1 + (t * 32 + lane) * 16);
#pragma unroll
    for (int m = 0; m < 2; ++m) {
      v8f c = {};
      c = __builtin_amdgcn_wmma_f32_16x16x32_f16(false, A1, false,
                                                 (m ? B1b.v : B1a.v),
                                                 (short)0, c, false, false);
#pragma unroll
      for (int r = 0; r < 4; ++r)
        hpk[t][m][r] = pk16(fmaxf(c[2 * r], 0.0f), fmaxf(c[2 * r + 1], 0.0f));
    }
  }

  // ---- layer 2: d_pre = W2^T h + b2; 2 K-tiles x 2 point-tiles ------------
  v8f c2a = *(const v8f*)(b2p + (low ? 0 : 8));   // C-init = b2[M], M=j+hi8
  v8f c2b = c2a;
#pragma unroll
  for (int kt = 0; kt < 2; ++kt) {
    v16h A2 = *(const v16h*)(Ap2 + (kt * 32 + lane) * 16);
#pragma unroll
    for (int m = 0; m < 2; ++m) {
      BOp B2;
#pragma unroll
      for (int r = 0; r < 4; ++r) {
        unsigned own0 = hpk[2 * kt][m][r];        // units 0-7 (low lanes own)
        unsigned own1 = hpk[2 * kt + 1][m][r];    // units 8-15 (high lanes own)
        unsigned sh0  = shx16u(own0);
        unsigned sh1  = shx16u(own1);
        B2.u[r]     = low ? own0 : sh1;           // K halfs 0-7
        B2.u[4 + r] = low ? sh0  : own1;          // K halfs 8-15
      }
      if (m == 0)
        c2a = __builtin_amdgcn_wmma_f32_16x16x32_f16(false, A2, false, B2.v,
                                                     (short)0, c2a, false, false);
      else
        c2b = __builtin_amdgcn_wmma_f32_16x16x32_f16(false, A2, false, B2.v,
                                                     (short)0, c2b, false, false);
    }
  }

  // ---- gather this lane's point: d[0..9] ----------------------------------
  // C2[m]: lane holds outputs j+(low?0:8) of point m*16 + lane%16.
  // low lane L  -> point L    (m=0): own d0..7, partner gives d8,d9
  // high lane L -> point L    (m=1): own d8..15, partner gives d0..7
  float sh_a0 = shx16f(c2a[0]);
  float sh_a1 = shx16f(c2a[1]);
  float sh_b[8];
#pragma unroll
  for (int j = 0; j < 8; ++j) sh_b[j] = shx16f(c2b[j]);

  float d[10];
#pragma unroll
  for (int j = 0; j < 8; ++j) d[j] = low ? c2a[j] : sh_b[j];
  d[8] = low ? sh_a0 : c2b[0];
  d[9] = low ? sh_a1 : c2b[1];

  // softplus (stable) + 1e-4
#pragma unroll
  for (int i = 0; i < 10; ++i) {
    float v = d[i];
    d[i] = fmaxf(v, 0.0f) + __logf(1.0f + __expf(-fabsf(v))) + 1.0e-4f;
  }

  // ---- piecewise-linear interpolation -------------------------------------
  float dxl2 = d[0], dxl1 = d[1], dxr1 = d[2], dxr2 = d[3];
  float dyl2 = d[4], dyl1 = d[5], dyr1 = d[6], dyr2 = d[7];
  float kl = 2.0f * d[8], kr = 2.0f * d[9];

  float xL1 = -dxl1,        yL1 = -dyl1;
  float xL2 = -dxl1 - dxl2, yL2 = -dyl1 - dyl2;
  float xR1 = dxr1,         yR1 = dyr1;
  float xR2 = dxr1 + dxr2,  yR2 = dyr1 + dyr2;
  float xR3 = xR2 + BIG,    yR3 = yR2 + kr * BIG;
  float xL3 = xL2 - BIG,    yL3 = yL2 - kl * BIG;

  float ax[6] = {xL3, xL2, xL1, xR1, xR2, xR3};
  float ay[6] = {yL3, yL2, yL1, yR1, yR2, yR3};

  float qx = fminf(fmaxf(x2, ax[0] * 0.99f), ax[5] * 0.99f);
  float s = 0.0f;
#pragma unroll
  for (int i = 0; i < 5; ++i) {
    float xl_ = ax[i], xr_ = ax[i + 1];
    float yl_ = ay[i], yr_ = ay[i + 1];
    bool in_range = (qx >= xl_) && (qx < xr_);
    float val = __fdividef(yr_ - yl_, xr_ - xl_) * (qx - xl_) + yl_;
    s += in_range ? val : 0.0f;
  }

  float* op = out + (size_t)p * 3;
  op[0] = x0 * mask[0];
  op[1] = x1 * mask[1];
  op[2] = s;
}

// ---------------------------------------------------------------------------
extern "C" void kernel_launch(void* const* d_in, const int* in_sizes, int n_in,
                              void* d_out, int out_size, void* d_ws, size_t ws_size,
                              hipStream_t stream) {
  const float* x      = (const float*)d_in[0];
  const float* t_feat = (const float*)d_in[1];
  const float* mask   = (const float*)d_in[2];
  const float* W1     = (const float*)d_in[3];
  const float* b1     = (const float*)d_in[4];
  const float* W2     = (const float*)d_in[5];
  const float* b2     = (const float*)d_in[6];
  float* out          = (float*)d_out;

  const int n = in_sizes[0] / 3;        // 2^21 points, divisible by 256

  _Float16* Ap1 = (_Float16*)d_ws;                     // 4 KB
  _Float16* Ap2 = (_Float16*)((char*)d_ws + 4096);     // 2 KB
  float*    b2p = (float*)((char*)d_ws + 6144);        // 64 B

  pack_weights_kernel<<<8, 256, 0, stream>>>(W1, b1, W2, b2, Ap1, Ap2, b2p);

  const int blocks = n / (32 * WAVES_PER_BLOCK);       // 256 points per block
  coupling_wmma_kernel<<<blocks, 256, 0, stream>>>(x, t_feat, mask,
                                                   Ap1, Ap2, b2p, out);
}
